// SpatialAttentionLSTMDecoder_8650064134706
// MI455X (gfx1250) — matrix operand order repaired and checked by reading, not compiled, not run on hardware
//
#include <hip/hip_runtime.h>
#include <hip/hip_bf16.h>

// ---------------------------------------------------------------------------
// SpatialAttentionLSTMDecoder for MI455X (gfx1250, wave32, WMMA bf16)
// V=10000, F=768, H=768, R=49, B=128, T=32
// ---------------------------------------------------------------------------

#define V_DIM 10000
#define F_DIM 768
#define H_DIM 768
#define R_DIM 49
#define B_DIM 128
#define T_DIM 32
#define G_DIM (4 * H_DIM)            // 3072
#define KCAT (H_DIM + F_DIM + H_DIM) // 2304 : [emb | context | h]

typedef __attribute__((ext_vector_type(16))) __bf16 v16bf;
typedef __attribute__((ext_vector_type(8)))  __bf16 v8bf;
typedef __attribute__((ext_vector_type(8)))  float  v8f;

__device__ __forceinline__ float sigm(float x) { return 1.0f / (1.0f + expf(-x)); }
__device__ __forceinline__ __bf16 to_bf16(float x) { return (__bf16)x; }

// ---------------------------------------------------------------------------
// Weight pre-swizzle into WMMA B-fragment order (bf16), padded to NT4 tiles
// (NT4 multiple of 4) with zeros so the GEMM inner loop needs no guards.
// Logical B[k][n] = W[n][coloff + k] (i.e. x @ W^T GEMM).
// Fragment order: out[(((kc*NT4 + nt)*32 + lane)*16) + e], lane = half*16+idx,
//   n = nt*16 + idx, k = kc*32 + half*16 + e     (contiguous k order, same as the GEMM A loads)
// ---------------------------------------------------------------------------
__global__ void __launch_bounds__(256)
pack_b_kernel(const float* __restrict__ W, int ldw, int coloff,
              int K, int N, int NT4, __bf16* __restrict__ out, long total) {
    long i = (long)blockIdx.x * 256 + threadIdx.x;
    if (i >= total) return;
    int e    = (int)(i & 15);
    int lane = (int)((i >> 4) & 31);
    long tile = i >> 9;
    int nt = (int)(tile % NT4);
    int kc = (int)(tile / NT4);
    int half = lane >> 4, idx = lane & 15;
    int n = nt * 16 + idx;
    int k = kc * 32 + half * 16 + e;
    float v = (n < N && k < K) ? W[(long)n * ldw + coloff + k] : 0.0f;
    out[i] = to_bf16(v);
}

// Same, but B rows are the concatenation [W_ih (K=1536) | W_hh (K=768)], N=3072.
__global__ void __launch_bounds__(256)
pack_wcat_kernel(const float* __restrict__ W_ih, const float* __restrict__ W_hh,
                 int NT4, __bf16* __restrict__ out, long total) {
    long i = (long)blockIdx.x * 256 + threadIdx.x;
    if (i >= total) return;
    int e    = (int)(i & 15);
    int lane = (int)((i >> 4) & 31);
    long tile = i >> 9;
    int nt = (int)(tile % NT4);
    int kc = (int)(tile / NT4);
    int half = lane >> 4, idx = lane & 15;
    int n = nt * 16 + idx;
    int k = kc * 32 + half * 16 + e;
    float v;
    if (k < 1536) v = W_ih[(long)n * 1536 + k];
    else          v = W_hh[(long)n * 768 + (k - 1536)];
    out[i] = to_bf16(v);
}

// ---------------------------------------------------------------------------
// bf16 WMMA GEMM:  C[M,N] = A[M,K](bf16, row-major, lda) @ Bpacked + bias
// Block: 256 threads = 8 waves; wave w -> rows [blockIdx.y*128 + w*16, +16),
// cols [blockIdx.x*64, +64) as 4 accumulators sharing the A fragment.
// Software-pipelined: fragments for kc+1 are loaded before the WMMAs of kc
// (reads one K-chunk of slack past the end; buffers over-allocated).
// ---------------------------------------------------------------------------
__global__ void __launch_bounds__(256)
gemm_bf16_wmma(const __bf16* __restrict__ A, int lda,
               const __bf16* __restrict__ Bp, int NT4,
               const float* __restrict__ bias,
               float* __restrict__ C, int ldc,
               int M, int N, int K) {
    const int wave = threadIdx.x >> 5;
    const int lane = threadIdx.x & 31;
    const int half = lane >> 4;
    const int idx  = lane & 15;
    const int m0 = blockIdx.y * 128 + wave * 16;
    const int n0 = blockIdx.x * 64;

    const int KC = K >> 5;
    const long bstride = (long)NT4 * 512;   // packed elements per K-chunk

    const __bf16* ap = A + (long)(m0 + idx) * lda + half * 16;
    const __bf16* bp = Bp + (((long)(n0 >> 4)) * 32 + lane) * 16;

    v8f acc0 = {}, acc1 = {}, acc2 = {}, acc3 = {};

    // prologue: fragments for kc = 0
    v8bf  alo = *(const v8bf*)(ap);
    v8bf  ahi = *(const v8bf*)(ap + 8);
    v16bf b0  = *(const v16bf*)(bp);
    v16bf b1  = *(const v16bf*)(bp + 512);
    v16bf b2  = *(const v16bf*)(bp + 1024);
    v16bf b3  = *(const v16bf*)(bp + 1536);

#pragma unroll 2
    for (int kc = 0; kc < KC; ++kc) {
        v16bf a;
#pragma unroll
        for (int e = 0; e < 8; ++e) { a[e] = alo[e]; a[e + 8] = ahi[e]; }
        v16bf c0 = b0, c1 = b1, c2 = b2, c3 = b3;

        // issue next-iteration loads before the WMMAs (double-buffered regs)
        ap += 32;
        bp += bstride;
        __builtin_prefetch(ap + 32, 0, 1);   // unconditional, speculative
        alo = *(const v8bf*)(ap);
        ahi = *(const v8bf*)(ap + 8);
        b0  = *(const v16bf*)(bp);
        b1  = *(const v16bf*)(bp + 512);
        b2  = *(const v16bf*)(bp + 1024);
        b3  = *(const v16bf*)(bp + 1536);

        acc0 = __builtin_amdgcn_wmma_f32_16x16x32_bf16(false, a, false, c0, (short)0, acc0, false, false);
        acc1 = __builtin_amdgcn_wmma_f32_16x16x32_bf16(false, a, false, c1, (short)0, acc1, false, false);
        acc2 = __builtin_amdgcn_wmma_f32_16x16x32_bf16(false, a, false, c2, (short)0, acc2, false, false);
        acc3 = __builtin_amdgcn_wmma_f32_16x16x32_bf16(false, a, false, c3, (short)0, acc3, false, false);
    }

    v8f accs[4] = {acc0, acc1, acc2, acc3};
#pragma unroll
    for (int j = 0; j < 4; ++j) {
        int n = n0 + j * 16 + idx;
        if (n < N) {
            float bv = bias ? bias[n] : 0.0f;
#pragma unroll
            for (int i = 0; i < 8; ++i) {
                C[(long)(m0 + i + half * 8) * ldc + n] = accs[j][i] + bv;
            }
        }
    }
}

// ---------------------------------------------------------------------------
// Small prep kernels
// ---------------------------------------------------------------------------
__global__ void __launch_bounds__(256)
bias_sum_kernel(const float* __restrict__ b_ih, const float* __restrict__ b_hh,
                float* __restrict__ out) {
    int i = blockIdx.x * 256 + threadIdx.x;
    if (i < G_DIM) out[i] = b_ih[i] + b_hh[i];
}

__global__ void __launch_bounds__(256)
f32_to_bf16_kernel(const float* __restrict__ in, __bf16* __restrict__ out, long n) {
    long i = (long)blockIdx.x * 256 + threadIdx.x;
    if (i < n) out[i] = to_bf16(in[i]);
}

__global__ void __launch_bounds__(256)
embed_kernel(const int* __restrict__ captions, const float* __restrict__ emb_W,
             __bf16* __restrict__ emb_bf) {
    long i = (long)blockIdx.x * 256 + threadIdx.x;
    if (i >= (long)B_DIM * T_DIM * H_DIM) return;
    int hh = (int)(i % H_DIM);
    long bt = i / H_DIM;
    int cap = captions[bt];
    emb_bf[i] = to_bf16(emb_W[(long)cap * H_DIM + hh]);
}

__global__ void __launch_bounds__(256)
init_h0_kernel(const float* __restrict__ features,
               float* __restrict__ h, float* __restrict__ c,
               __bf16* __restrict__ h_bf) {
    int i = blockIdx.x * 256 + threadIdx.x;
    if (i >= B_DIM * H_DIM) return;
    int b = i / H_DIM, j = i % H_DIM;
    const float* fb = features + (long)b * R_DIM * F_DIM + j;
    float s = 0.0f;
    for (int r = 0; r < R_DIM; ++r) s += fb[r * F_DIM];
    float hv = tanhf(s * (1.0f / (float)R_DIM));
    h[i] = hv; c[i] = 0.0f; h_bf[i] = to_bf16(hv);
}

// ---------------------------------------------------------------------------
// Fused attention: scores -> softmax over R -> context; assembles bf16 GEMM
// input row xh = [emb_t | context | h].  One block per batch element.
// ---------------------------------------------------------------------------
__global__ void __launch_bounds__(256)
attn_context_kernel(const float* __restrict__ feat_proj,   // (B*R, H), attn_b baked in
                    const float* __restrict__ hWh,         // (B, H)
                    const float* __restrict__ features,    // (B, R, F) fp32
                    const float* __restrict__ vvec,        // (H) = attnv_W[0]
                    const __bf16* __restrict__ emb_bf,     // (B, T, H)
                    const __bf16* __restrict__ h_bf,       // (B, H)
                    __bf16* __restrict__ xh,               // (B, KCAT)
                    int t) {
    __shared__ float shWh[H_DIM];
    __shared__ float sc[64];
    const int b = blockIdx.x;
    const int tid = threadIdx.x;
    const int lane = tid & 31;
    const int w = tid >> 5;

    for (int i = tid; i < H_DIM; i += 256) shWh[i] = hWh[(long)b * H_DIM + i];
    __syncthreads();

    const float* fp = feat_proj + (long)b * R_DIM * H_DIM;
    for (int r = w; r < R_DIM; r += 8) {
        float s = 0.0f;
        for (int hh = lane; hh < H_DIM; hh += 32)
            s += tanhf(fp[r * H_DIM + hh] + shWh[hh]) * vvec[hh];
#pragma unroll
        for (int off = 16; off > 0; off >>= 1) s += __shfl_down(s, off, 32);
        if (lane == 0) sc[r] = s;
    }
    __syncthreads();

    if (w == 0) {  // softmax over 49 scores (attnv_b shift cancels in softmax)
        float m = -1e30f;
        for (int r = lane; r < R_DIM; r += 32) m = fmaxf(m, sc[r]);
#pragma unroll
        for (int off = 16; off > 0; off >>= 1) m = fmaxf(m, __shfl_down(m, off, 32));
        m = __shfl(m, 0, 32);
        float s = 0.0f;
        for (int r = lane; r < R_DIM; r += 32) { float e = expf(sc[r] - m); sc[r] = e; s += e; }
#pragma unroll
        for (int off = 16; off > 0; off >>= 1) s += __shfl_down(s, off, 32);
        s = __shfl(s, 0, 32);
        float inv = 1.0f / s;
        for (int r = lane; r < R_DIM; r += 32) sc[r] *= inv;
    }
    __syncthreads();

    const float* feat_b = features + (long)b * R_DIM * F_DIM;
    const __bf16* embrow = emb_bf + ((long)b * T_DIM + t) * H_DIM;
    const __bf16* hrow = h_bf + (long)b * H_DIM;
    __bf16* xrow = xh + (long)b * KCAT;
    for (int f = tid; f < F_DIM; f += 256) {
        float cacc = 0.0f;
        for (int r = 0; r < R_DIM; ++r) cacc += sc[r] * feat_b[r * F_DIM + f];
        xrow[f]                 = embrow[f];
        xrow[H_DIM + f]         = to_bf16(cacc);
        xrow[H_DIM + F_DIM + f] = hrow[f];
    }
}

// ---------------------------------------------------------------------------
// LSTM pointwise update
// ---------------------------------------------------------------------------
__global__ void __launch_bounds__(256)
lstm_point_kernel(const float* __restrict__ gates,
                  float* __restrict__ h, float* __restrict__ c,
                  __bf16* __restrict__ h_bf) {
    int i = blockIdx.x * 256 + threadIdx.x;
    if (i >= B_DIM * H_DIM) return;
    int b = i / H_DIM, j = i % H_DIM;
    const float* g = gates + (long)b * G_DIM;
    float ig = sigm(g[j]);
    float fg = sigm(g[H_DIM + j]);
    float gg = tanhf(g[2 * H_DIM + j]);
    float og = sigm(g[3 * H_DIM + j]);
    float cn = fg * c[i] + ig * gg;
    float hn = og * tanhf(cn);
    c[i] = cn; h[i] = hn; h_bf[i] = to_bf16(hn);
}

__global__ void __launch_bounds__(256)
copy_hc_kernel(const float* __restrict__ h, const float* __restrict__ c,
               float* __restrict__ out_tail) {
    int i = blockIdx.x * 256 + threadIdx.x;
    int n = B_DIM * H_DIM;
    if (i < n)           out_tail[i] = h[i];
    else if (i < 2 * n)  out_tail[i] = c[i - n];
}

// ---------------------------------------------------------------------------
// Host launcher
// ---------------------------------------------------------------------------
static inline void* ws_take(char*& p, size_t bytes) {
    void* r = (void*)p;
    p += (bytes + 255) & ~(size_t)255;
    return r;
}

extern "C" void kernel_launch(void* const* d_in, const int* in_sizes, int n_in,
                              void* d_out, int out_size, void* d_ws, size_t ws_size,
                              hipStream_t stream) {
    const float* features = (const float*)d_in[0];
    const int*   captions = (const int*)d_in[1];
    const float* emb_W    = (const float*)d_in[2];
    const float* attn_W   = (const float*)d_in[3];
    const float* attn_b   = (const float*)d_in[4];
    const float* attnv_W  = (const float*)d_in[5];
    // d_in[6] attnv_b: constant shift before softmax -> no effect, skipped
    const float* W_ih     = (const float*)d_in[7];
    const float* W_hh     = (const float*)d_in[8];
    const float* b_ih     = (const float*)d_in[9];
    const float* b_hh     = (const float*)d_in[10];
    const float* fc_W     = (const float*)d_in[11];
    const float* fc_b     = (const float*)d_in[12];
    float* out = (float*)d_out;

    // ---- workspace carve (all 256B aligned) ----
    char* p = (char*)d_ws;
    // padded N-tile counts (multiples of 4 so the GEMM strip is unguarded)
    const int NT4_H = H_DIM / 16;                   // 48  (already mult of 4)
    const int NT4_G = G_DIM / 16;                   // 192
    const int NT4_V = ((V_DIM + 63) / 64) * 4;      // 628
    const int KC_H   = H_DIM / 32;                  // 24
    const int KC_CAT = KCAT / 32;                   // 72

    // packed weights get one extra K-chunk (NT4*512 elems) of pipeline slack;
    // bf16 activation buffers get 128 B of slack for the same reason.
    __bf16* WfT_pack  = (__bf16*)ws_take(p, (size_t)(KC_H + 1) * NT4_H * 512 * 2);
    __bf16* WhT_pack  = (__bf16*)ws_take(p, (size_t)(KC_H + 1) * NT4_H * 512 * 2);
    __bf16* Wcat_pack = (__bf16*)ws_take(p, (size_t)(KC_CAT + 1) * NT4_G * 512 * 2);
    __bf16* fcT_pack  = (__bf16*)ws_take(p, (size_t)(KC_H + 1) * NT4_V * 512 * 2);
    __bf16* feats_bf  = (__bf16*)ws_take(p, (size_t)B_DIM * R_DIM * F_DIM * 2 + 128);
    __bf16* emb_bf    = (__bf16*)ws_take(p, (size_t)B_DIM * T_DIM * H_DIM * 2);
    float*  feat_proj = (float*)ws_take(p, (size_t)B_DIM * R_DIM * H_DIM * 4);
    float*  hWh       = (float*)ws_take(p, (size_t)B_DIM * H_DIM * 4);
    __bf16* xh        = (__bf16*)ws_take(p, (size_t)B_DIM * KCAT * 2 + 128);
    float*  gates     = (float*)ws_take(p, (size_t)B_DIM * G_DIM * 4);
    float*  h_st      = (float*)ws_take(p, (size_t)B_DIM * H_DIM * 4);
    float*  c_st      = (float*)ws_take(p, (size_t)B_DIM * H_DIM * 4);
    __bf16* h_bf      = (__bf16*)ws_take(p, (size_t)B_DIM * H_DIM * 2 + 128);
    float*  bsum      = (float*)ws_take(p, (size_t)G_DIM * 4);

    // ---- one-time prep ----
    {
        long tot = (long)KC_H * NT4_H * 512;
        pack_b_kernel<<<dim3((tot + 255) / 256), 256, 0, stream>>>(
            attn_W, F_DIM + H_DIM, 0, H_DIM, H_DIM, NT4_H, WfT_pack, tot);
        pack_b_kernel<<<dim3((tot + 255) / 256), 256, 0, stream>>>(
            attn_W, F_DIM + H_DIM, F_DIM, H_DIM, H_DIM, NT4_H, WhT_pack, tot);
        long totc = (long)KC_CAT * NT4_G * 512;
        pack_wcat_kernel<<<dim3((totc + 255) / 256), 256, 0, stream>>>(
            W_ih, W_hh, NT4_G, Wcat_pack, totc);
        long totf = (long)KC_H * NT4_V * 512;
        pack_b_kernel<<<dim3((totf + 255) / 256), 256, 0, stream>>>(
            fc_W, H_DIM, 0, H_DIM, V_DIM, NT4_V, fcT_pack, totf);
    }
    bias_sum_kernel<<<dim3((G_DIM + 255) / 256), 256, 0, stream>>>(b_ih, b_hh, bsum);
    {
        long n = (long)B_DIM * R_DIM * F_DIM;
        f32_to_bf16_kernel<<<dim3((n + 255) / 256), 256, 0, stream>>>(features, feats_bf, n);
        long ne = (long)B_DIM * T_DIM * H_DIM;
        embed_kernel<<<dim3((ne + 255) / 256), 256, 0, stream>>>(captions, emb_W, emb_bf);
    }
    init_h0_kernel<<<dim3((B_DIM * H_DIM + 255) / 256), 256, 0, stream>>>(
        features, h_st, c_st, h_bf);

    // feat_proj = features @ Wf^T + attn_b   (M=6272, N=768, K=768)
    gemm_bf16_wmma<<<dim3(NT4_H / 4, (B_DIM * R_DIM) / 128), 256, 0, stream>>>(
        feats_bf, F_DIM, WfT_pack, NT4_H, attn_b,
        feat_proj, H_DIM, B_DIM * R_DIM, H_DIM, F_DIM);

    // ---- 32 sequential decode steps ----
    for (int t = 0; t < T_DIM; ++t) {
        // hWh = h @ Wh^T           (M=128, N=768, K=768)
        gemm_bf16_wmma<<<dim3(NT4_H / 4, 1), 256, 0, stream>>>(
            h_bf, H_DIM, WhT_pack, NT4_H, nullptr,
            hWh, H_DIM, B_DIM, H_DIM, H_DIM);

        // attention scores/softmax/context + assemble xh = [emb_t|ctx|h]
        attn_context_kernel<<<dim3(B_DIM), 256, 0, stream>>>(
            feat_proj, hWh, features, attnv_W, emb_bf, h_bf, xh, t);

        // gates = xh @ [W_ih|W_hh]^T + (b_ih+b_hh)   (M=128, N=3072, K=2304)
        gemm_bf16_wmma<<<dim3(NT4_G / 4, 1), 256, 0, stream>>>(
            xh, KCAT, Wcat_pack, NT4_G, bsum,
            gates, G_DIM, B_DIM, G_DIM, KCAT);

        // LSTM cell pointwise
        lstm_point_kernel<<<dim3((B_DIM * H_DIM + 255) / 256), 256, 0, stream>>>(
            gates, h_st, c_st, h_bf);

        // logits_t = h_new @ fc_W^T + fc_b   (M=128, N=10000, K=768)
        // write directly into d_out[b][t][:], ldc = T*V
        gemm_bf16_wmma<<<dim3(NT4_V / 4, 1), 256, 0, stream>>>(
            h_bf, H_DIM, fcT_pack, NT4_V, fc_b,
            out + (long)t * V_DIM, T_DIM * V_DIM, B_DIM, V_DIM, H_DIM);
    }

    // final (h, c) after outputs
    copy_hc_kernel<<<dim3((2 * B_DIM * H_DIM + 255) / 256), 256, 0, stream>>>(
        h_st, c_st, out + (long)B_DIM * T_DIM * V_DIM);
}
